// AMRLMax_26027501814271
// MI455X (gfx1250) — matrix operand order repaired and checked by reading, not compiled, not run on hardware
//
#include <hip/hip_runtime.h>
#include <hip/hip_bf16.h>
#include <stdint.h>

// Problem dims (match reference)
#define B_    64
#define T_    512
#define I_    256
#define H_    512
#define HH_   256
#define G4H   2048   // 4*H

typedef __attribute__((ext_vector_type(16))) __bf16 v16bf;
typedef __attribute__((ext_vector_type(8)))  float  v8f;
typedef __attribute__((ext_vector_type(4)))  unsigned int v4u;
typedef __attribute__((ext_vector_type(8)))  int v8i_;
typedef __attribute__((ext_vector_type(4)))  int v4i_;

#if __has_builtin(__builtin_amdgcn_tensor_load_to_lds)
#define HAS_TDM 1
#else
#define HAS_TDM 0
#endif

__device__ __forceinline__ unsigned short f2bf(float f) {
  union { float f; unsigned u; } x; x.f = f;
  unsigned u = x.u;
  u += 0x7FFFu + ((u >> 16) & 1u);     // round-to-nearest-even
  return (unsigned short)(u >> 16);
}

union FragBF { v16bf v; uint4 q[2]; };

// A fragment: 16x32 bf16 tile, rows m=0..15 relative to `base`, cols k0..k0+31.
__device__ __forceinline__ v16bf load_frag_a(const unsigned short* base, int lda, int k0) {
  const int lane = threadIdx.x & 31;
  const int m = lane & 15, kh = lane >> 4;
  FragBF f;
  const unsigned short* p = base + (size_t)m * lda + (k0 + kh * 8);
  f.q[0] = *(const uint4*)(p);
  f.q[1] = *(const uint4*)(p + 16);
  return f.v;
}

// B fragment: 32x16 where B[k][n] = W[n][k]; W row-major, rows n..n+15 at `base`.
__device__ __forceinline__ v16bf load_frag_b(const unsigned short* base, int ldw, int k0) {
  const int lane = threadIdx.x & 31;
  const int n = lane & 15, kh = lane >> 4;
  FragBF f;
  const unsigned short* p = base + (size_t)n * ldw + (k0 + kh * 8);
  f.q[0] = *(const uint4*)(p);
  f.q[1] = *(const uint4*)(p + 16);
  return f.v;
}

__device__ __forceinline__ v8f wmma_bf16(v16bf a, v16bf b, v8f c) {
  return __builtin_amdgcn_wmma_f32_16x16x32_bf16(false, a, false, b, (short)0, c, false, false);
}

__device__ __forceinline__ float sigm(float x) { return 1.0f / (1.0f + __expf(-x)); }

// ---------------------------------------------------------------------------
// Prep: convert weights to bf16, fold biases, init h0, zero barrier ctrl.
// ---------------------------------------------------------------------------
__global__ void k_prep(const float* __restrict__ Wih, const float* __restrict__ Whh,
                       const float* __restrict__ bih, const float* __restrict__ bhh,
                       const float* __restrict__ mem,
                       unsigned short* __restrict__ Wih_bf, unsigned short* __restrict__ Whh_bf,
                       float* __restrict__ bias, unsigned short* __restrict__ hbf0,
                       unsigned* __restrict__ ctrl) {
  size_t tid = (size_t)blockIdx.x * blockDim.x + threadIdx.x;
  size_t stride = (size_t)gridDim.x * blockDim.x;
  if (tid < 2) ctrl[tid] = 0u;
  for (size_t i = tid; i < (size_t)G4H * H_; i += stride) Whh_bf[i] = f2bf(Whh[i]);
  for (size_t i = tid; i < (size_t)G4H * I_; i += stride) Wih_bf[i] = f2bf(Wih[i]);
  for (size_t i = tid; i < (size_t)G4H; i += stride) bias[i] = bih[i] + bhh[i];
  for (size_t i = tid; i < (size_t)B_ * H_; i += stride) {
    size_t b = i >> 9, j = i & 511;
    hbf0[i] = f2bf(mem[b * 1280 + 256 + j]);   // h0
  }
}

// Xbf[t][b][i] = bf16(X[b][t][i])
__global__ void k_xcast(const float* __restrict__ X, unsigned short* __restrict__ Xbf) {
  size_t tid = (size_t)blockIdx.x * blockDim.x + threadIdx.x;
  size_t stride = (size_t)gridDim.x * blockDim.x;
  const size_t N = (size_t)T_ * B_ * I_;
  for (size_t idx = tid; idx < N; idx += stride) {
    size_t i = idx % I_;
    size_t b = (idx / I_) % B_;
    size_t t = idx / ((size_t)I_ * B_);
    Xbf[idx] = f2bf(X[((size_t)b * T_ + t) * I_ + i]);
  }
}

// ---------------------------------------------------------------------------
// Phase 1: Gx = Xbf(32768x256) @ Wih^T + bias -> (32768x2048) f32.
// ---------------------------------------------------------------------------
__global__ void k_gemm_x(const unsigned short* __restrict__ Xbf,
                         const unsigned short* __restrict__ Wih_bf,
                         const float* __restrict__ bias, float* __restrict__ Gx) {
  const int wave = threadIdx.x >> 5;
  const int mstrip = wave & 3, nhalf = wave >> 2;
  const int row0 = blockIdx.x * 64 + mstrip * 16;
  const int colbase = blockIdx.y * 64 + nhalf * 32;

  v8f acc0 = {}, acc1 = {};
  const unsigned short* A  = Xbf + (size_t)row0 * I_;
  const unsigned short* B0 = Wih_bf + (size_t)colbase * I_;
  const unsigned short* B1 = Wih_bf + (size_t)(colbase + 16) * I_;
  for (int k0 = 0; k0 < I_; k0 += 32) {
    v16bf a  = load_frag_a(A, I_, k0);
    acc0 = wmma_bf16(a, load_frag_b(B0, I_, k0), acc0);
    acc1 = wmma_bf16(a, load_frag_b(B1, I_, k0), acc1);
  }
  const int lane = threadIdx.x & 31, nC = lane & 15, khC = lane >> 4;
  for (int r = 0; r < 8; ++r) {
    int row = row0 + khC * 8 + r;
    int c0i = colbase + nC;
    Gx[(size_t)row * G4H + c0i]      = acc0[r] + bias[c0i];
    Gx[(size_t)row * G4H + c0i + 16] = acc1[r] + bias[c0i + 16];
  }
}

// ---------------------------------------------------------------------------
// Device-wide barrier (16 co-resident WGs, one per WGP).
// ---------------------------------------------------------------------------
__device__ __forceinline__ void grid_barrier(unsigned* bar, unsigned* arrive, unsigned nwg) {
  __syncthreads();
  if (threadIdx.x == 0) {
    __threadfence();
    unsigned gen  = __hip_atomic_load(bar, __ATOMIC_ACQUIRE, __HIP_MEMORY_SCOPE_AGENT);
    unsigned prev = __hip_atomic_fetch_add(arrive, 1u, __ATOMIC_ACQ_REL, __HIP_MEMORY_SCOPE_AGENT);
    if (prev == nwg - 1) {
      __hip_atomic_store(arrive, 0u, __ATOMIC_RELAXED, __HIP_MEMORY_SCOPE_AGENT);
      __hip_atomic_fetch_add(bar, 1u, __ATOMIC_ACQ_REL, __HIP_MEMORY_SCOPE_AGENT);
    } else {
      while (__hip_atomic_load(bar, __ATOMIC_ACQUIRE, __HIP_MEMORY_SCOPE_AGENT) == gen) {
        __builtin_amdgcn_s_sleep(2);
      }
    }
  }
  __syncthreads();
}

// ---------------------------------------------------------------------------
// TDM: one tensor_load_to_lds per WG per step stages the 32KB Gx slice.
// Tile: x = 32 cols (contig), y = 4 gates (stride 512 f32), z = 64 batch
// (stride 2048 f32). LDS result layout = [b][gate][32] contiguous.
// D# packing per CDNA5 ISA (groups 0..3); this
// toolchain's builtin takes 6 args (extra int32x8 before cpol) -> zero-filled.
// ---------------------------------------------------------------------------
__device__ __forceinline__ void tdm_stage_gx(float* lds_dst, const float* gsrc) {
#if HAS_TDM
  unsigned lds_off = (unsigned)(uintptr_t)(__attribute__((address_space(3))) float*)lds_dst;
  unsigned long long ga = (unsigned long long)(uintptr_t)gsrc;
  v4u g0;
  g0[0] = 1u;                                   // count=1, user descriptor
  g0[1] = lds_off;                              // lds_addr
  g0[2] = (unsigned)ga;                         // global_addr[31:0]
  g0[3] = (unsigned)((ga >> 32) & 0x1FFFFFFu) | (2u << 30);  // addr[56:32] | type=2
  v8i_ g1;
  g1[0] = (int)(2u << 16);                      // workgroup_mask=0, data_size=4B
  g1[1] = (int)(2048u << 16);                   // tensor_dim0 = 2048 (low 16 into [31:16])
  g1[2] = (int)(4u << 16);                      // tensor_dim1 = 4
  g1[3] = (int)(32u << 16);                     // tile_dim0 = 32
  g1[4] = (int)(4u | (64u << 16));              // tile_dim1 = 4, tile_dim2 = 64
  g1[5] = 512;                                  // tensor_dim0_stride[31:0] = 512
  g1[6] = (int)(2048u << 16);                   // tensor_dim1_stride[15:0] = 2048
  g1[7] = 0;
  v4i_ g2; g2[0] = 64; g2[1] = 1; g2[2] = 0; g2[3] = 0;   // tensor_dim2=64
  v4i_ g3; g3[0] = 0;  g3[1] = 0; g3[2] = 0; g3[3] = 0;
  v8i_ g4;                                      // zero-filled trailing group
  g4[0] = 0; g4[1] = 0; g4[2] = 0; g4[3] = 0;
  g4[4] = 0; g4[5] = 0; g4[6] = 0; g4[7] = 0;
  __builtin_amdgcn_tensor_load_to_lds(g0, g1, g2, g3, g4, 0);
#else
  (void)lds_dst; (void)gsrc;
#endif
}

__device__ __forceinline__ void tdm_wait() {
#if HAS_TDM
#if __has_builtin(__builtin_amdgcn_s_wait_tensorcnt)
  __builtin_amdgcn_s_wait_tensorcnt(0);
#endif
#endif
}

// Fallback synchronous stage (also used when TDM builtin is unavailable).
__device__ __forceinline__ void stage_gx_fallback(float* dst, const float* src_t, int wgcol) {
  int p = threadIdx.x;                 // 256 threads = 64 b x 4 gates
  int b = p >> 2, g = p & 3;
  const float* s = src_t + (size_t)b * G4H + g * 512 + wgcol;
  float* d = dst + p * 32;
  #pragma unroll
  for (int ch = 0; ch < 8; ++ch) *(uint4*)(d + ch * 4) = *(const uint4*)(s + ch * 4);
}

// ---------------------------------------------------------------------------
// Phase 2: persistent LSTM scan. 16 blocks x 256 threads; block j owns h-cols
// [32j, 32j+32). W_hh slice lives in LDS (staged once); Gx slice double-
// buffered in LDS via TDM one step ahead; c/cummax state lives in registers.
// ---------------------------------------------------------------------------
__global__ void k_lstm(const float* __restrict__ Gx, const unsigned short* __restrict__ Whh_bf,
                       const float* __restrict__ mem,
                       unsigned short* __restrict__ hbf0, unsigned short* __restrict__ hbf1,
                       float* __restrict__ out, unsigned* __restrict__ ctrl) {
  extern __shared__ char smem[];
  unsigned short* Wlds = (unsigned short*)smem;            // 128 rows x 512 bf16 = 128KB
  float* gx0 = (float*)(smem + 128 * 512 * 2);             // 2 x 32KB Gx double buffer
  float* gx1 = gx0 + 64 * 128;

  const int wave = threadIdx.x >> 5;
  const int mstrip = wave & 3;                             // 16-row batch strip
  const int nt = wave >> 2;                                // 16-col tile (0..1)
  const int wgcol = blockIdx.x * 32;
  const int lane = threadIdx.x & 31, nC = lane & 15, khC = lane >> 4;
  const int col = wgcol + nt * 16 + nC;                    // fixed h column per lane
  unsigned* bar = ctrl;
  unsigned* arrive = ctrl + 1;
  const unsigned nwg = gridDim.x;

  // --- Stage this WG's W_hh slice into LDS once: row = g*32 + c ---
  for (int i = threadIdx.x; i < 128 * 64; i += 256) {      // 16B chunks
    int row = i >> 6, ch = i & 63;
    int g = row >> 5, c = row & 31;
    *(uint4*)(Wlds + row * 512 + ch * 8) =
        *(const uint4*)(Whh_bf + (size_t)(g * 512 + wgcol + c) * 512 + ch * 8);
  }
  // --- Stage Gx slice for t=0 ---
#if HAS_TDM
  if (wave == 0) tdm_stage_gx(gx0, Gx + wgcol);
  tdm_wait();
#else
  stage_gx_fallback(gx0, Gx, wgcol);
#endif
  __syncthreads();

  // --- Register-resident cell state and running cummax ---
  float creg[8], mreg[8];
  #pragma unroll
  for (int r = 0; r < 8; ++r) {
    int b = mstrip * 16 + khC * 8 + r;
    creg[r] = mem[b * 1280 + 768 + col];                   // c0
    mreg[r] = (col < HH_) ? mem[b * 1280 + col] : 0.0f;    // cummax seed
  }

  for (int t = 0; t < T_; ++t) {
    const unsigned short* hsrc = (t & 1) ? hbf1 : hbf0;
    unsigned short*       hdst = (t & 1) ? hbf0 : hbf1;
    float* cur = (t & 1) ? gx1 : gx0;
    float* nxt = (t & 1) ? gx0 : gx1;

    if (t + 1 < T_) {                                      // stage next step's slice
#if HAS_TDM
      if (wave == 0) tdm_stage_gx(nxt, Gx + (size_t)(t + 1) * B_ * G4H + wgcol);
#else
      stage_gx_fallback(nxt, Gx + (size_t)(t + 1) * B_ * G4H, wgcol);
#endif
    }

    // gates = h @ Whh^T : A from global (L2-hot broadcast), B from LDS
    v8f acc[4] = {};
    const unsigned short* A = hsrc + (size_t)(mstrip * 16) * H_;
    for (int k0 = 0; k0 < H_; k0 += 32) {
      v16bf a = load_frag_a(A, H_, k0);
      #pragma unroll
      for (int g = 0; g < 4; ++g) {
        acc[g] = wmma_bf16(a, load_frag_b(Wlds + (size_t)(g * 32 + nt * 16) * 512, 512, k0),
                           acc[g]);
      }
    }

    #pragma unroll
    for (int r = 0; r < 8; ++r) {
      const int b = mstrip * 16 + khC * 8 + r;
      const int li = b * 128 + nt * 16 + nC;               // [b][gate][32] in LDS
      float gi = acc[0][r] + cur[li];
      float gf = acc[1][r] + cur[li + 32];
      float gg = acc[2][r] + cur[li + 64];
      float go = acc[3][r] + cur[li + 96];
      float cn = sigm(gf) * creg[r] + sigm(gi) * tanhf(gg);
      float hn = sigm(go) * tanhf(cn);
      creg[r] = cn;
      hdst[b * H_ + col] = f2bf(hn);
      float oval = hn;
      if (col < HH_) { mreg[r] = fmaxf(mreg[r], hn); oval = mreg[r]; }
      out[((size_t)b * T_ + t) * H_ + col] = oval;         // concat_out[b][t][col]
      if (t == T_ - 1) {
        float* om = out + (size_t)B_ * T_ * H_ + (size_t)b * (HH_ + 2 * H_);
        if (col < HH_) om[col] = mreg[r];                  // m_final
        om[HH_ + col] = hn;                                // hT
        om[HH_ + H_ + col] = cn;                           // cT
      }
    }

    grid_barrier(bar, arrive, nwg);
#if HAS_TDM
    tdm_wait();                                            // next slice resident
    __syncthreads();
#endif
  }
}

// ---------------------------------------------------------------------------
extern "C" void kernel_launch(void* const* d_in, const int* in_sizes, int n_in,
                              void* d_out, int out_size, void* d_ws, size_t ws_size,
                              hipStream_t stream) {
  (void)in_sizes; (void)n_in; (void)out_size; (void)ws_size;
  const float* X   = (const float*)d_in[0];  // (B,T,I)
  const float* mem = (const float*)d_in[1];  // (1,B,1280)
  const float* Wih = (const float*)d_in[2];  // (2048,256)
  const float* Whh = (const float*)d_in[3];  // (2048,512)
  const float* bih = (const float*)d_in[4];
  const float* bhh = (const float*)d_in[5];
  float* out = (float*)d_out;

  char* ws = (char*)d_ws;
  size_t off = 0;
  auto carve = [&](size_t bytes) -> char* {
    char* p = ws + off;
    off += (bytes + 255) & ~(size_t)255;
    return p;
  };
  unsigned*       ctrl   = (unsigned*)carve(2 * sizeof(unsigned));
  float*          bias   = (float*)carve((size_t)G4H * 4);
  unsigned short* Wih_bf = (unsigned short*)carve((size_t)G4H * I_ * 2);
  unsigned short* Whh_bf = (unsigned short*)carve((size_t)G4H * H_ * 2);
  unsigned short* hbf0   = (unsigned short*)carve((size_t)B_ * H_ * 2);
  unsigned short* hbf1   = (unsigned short*)carve((size_t)B_ * H_ * 2);
  unsigned short* Xbf    = (unsigned short*)carve((size_t)T_ * B_ * I_ * 2);
  float*          Gx     = (float*)carve((size_t)T_ * B_ * G4H * 4);

  k_prep<<<dim3(1024), dim3(256), 0, stream>>>(Wih, Whh, bih, bhh, mem,
                                               Wih_bf, Whh_bf, bias, hbf0, ctrl);
  k_xcast<<<dim3(8192), dim3(256), 0, stream>>>(X, Xbf);
  k_gemm_x<<<dim3(T_ * B_ / 64, G4H / 64), dim3(256), 0, stream>>>(Xbf, Wih_bf, bias, Gx);
  k_lstm<<<dim3(16), dim3(256), 196608, stream>>>(Gx, Whh_bf, mem, hbf0, hbf1, out, ctrl);
}
